// GPT2AttentionTP_50921132261392
// MI455X (gfx1250) — matrix operand order repaired and checked by reading, not compiled, not run on hardware
//
#include <hip/hip_runtime.h>

typedef unsigned short u16;
typedef unsigned int   u32;

typedef __attribute__((ext_vector_type(8)))  __bf16 v8bf;
typedef __attribute__((ext_vector_type(16))) __bf16 v16bf;
typedef __attribute__((ext_vector_type(8)))  float  v8f;

#define DEVINL __device__ __forceinline__

// fp32 -> bf16 bits, round-to-nearest-even
DEVINL u16 f2bf(float f) {
  u32 u = __float_as_uint(f);
  u32 r = u + 0x7FFFu + ((u >> 16) & 1u);
  return (u16)(r >> 16);
}
DEVINL u32 pack2(u16 lo, u16 hi) { return (u32)lo | ((u32)hi << 16); }

// Load a 16-element bf16 fragment as two aligned 16B LDS reads.
DEVINL v16bf ld_frag(const u16* base, int off0, int off1) {
  v8bf lo = *(const v8bf*)(base + off0);
  v8bf hi = *(const v8bf*)(base + off1);
  return __builtin_shufflevector(lo, hi, 0, 1, 2, 3, 4, 5, 6, 7, 8, 9, 10, 11, 12, 13, 14, 15);
}

DEVINL v8f wmma_bf16(v16bf a, v16bf b, v8f c) {
  // D(f32 16x16) = A(bf16 16x32) * B(bf16 32x16) + C
  return __builtin_amdgcn_wmma_f32_16x16x32_bf16(false, a, false, b, (short)0, c, false, false);
}

// ---------------------------------------------------------------------------
// C[M,N] = A[M,K] @ W[N,K]^T + bias[N]
// A: fp32 (converted to bf16 while staging) or bf16; W: fp32; C: bf16 or fp32.
// Block: 256 threads (8 waves), 128x128 C tile, 32-deep K steps.
// Software pipeline: global loads for step kt+1 are issued before the WMMA
// block of step kt, so HBM/L2 latency hides under matrix math.
// ---------------------------------------------------------------------------
template <bool A_BF16, bool OUT_BF16>
__global__ __launch_bounds__(256) void gemm_bias_wmma(const void* __restrict__ Ap,
                                                      const float* __restrict__ W,
                                                      const float* __restrict__ bias,
                                                      void* __restrict__ Cp,
                                                      int M, int N, int K) {
  __shared__ __align__(16) u16 alds[128 * 40];  // 128 rows x 32 k (+8 pad)
  __shared__ __align__(16) u16 wlds[128 * 40];

  const int tid = threadIdx.x;
  const int n0 = blockIdx.x * 128, m0 = blockIdx.y * 128;
  const int lane = tid & 31, wv = tid >> 5;
  const int ln = lane & 15, hi = lane >> 4;
  const int wm = (wv & 1) * 64, wn = (wv >> 1) * 32;

  const float* A32 = (const float*)Ap;
  const u16* A16 = (const u16*)Ap;

  v8f acc[4][2] = {};

  // prefetch registers for the next K-step
  float4 pa[4];
  uint4 pa16[2];
  float4 pw[4];

  auto gload = [&](int k0) {
    if constexpr (!A_BF16) {
#pragma unroll
      for (int i = 0; i < 4; ++i) {
        int vec = i * 256 + tid;  // 1024 float4 vectors over [128][32]
        int row = vec >> 3, c4 = (vec & 7) << 2;
        pa[i] = *(const float4*)(A32 + (size_t)(m0 + row) * K + k0 + c4);
      }
    } else {
#pragma unroll
      for (int i = 0; i < 2; ++i) {
        int vec = i * 256 + tid;  // 512 uint4 vectors (8 bf16 each)
        int row = vec >> 2, c8 = (vec & 3) << 3;
        pa16[i] = *(const uint4*)(A16 + (size_t)(m0 + row) * K + k0 + c8);
      }
    }
#pragma unroll
    for (int i = 0; i < 4; ++i) {
      int vec = i * 256 + tid;
      int row = vec >> 3, c4 = (vec & 7) << 2;
      pw[i] = *(const float4*)(W + (size_t)(n0 + row) * K + k0 + c4);
    }
  };

  auto lstore = [&]() {
    if constexpr (!A_BF16) {
#pragma unroll
      for (int i = 0; i < 4; ++i) {
        int vec = i * 256 + tid;
        int row = vec >> 3, c4 = (vec & 7) << 2;
        uint2 p;
        p.x = pack2(f2bf(pa[i].x), f2bf(pa[i].y));
        p.y = pack2(f2bf(pa[i].z), f2bf(pa[i].w));
        *(uint2*)(alds + row * 40 + c4) = p;
      }
    } else {
#pragma unroll
      for (int i = 0; i < 2; ++i) {
        int vec = i * 256 + tid;
        int row = vec >> 2, c8 = (vec & 3) << 3;
        *(uint4*)(alds + row * 40 + c8) = pa16[i];
      }
    }
#pragma unroll
    for (int i = 0; i < 4; ++i) {
      int vec = i * 256 + tid;
      int row = vec >> 3, c4 = (vec & 7) << 2;
      uint2 p;
      p.x = pack2(f2bf(pw[i].x), f2bf(pw[i].y));
      p.y = pack2(f2bf(pw[i].z), f2bf(pw[i].w));
      *(uint2*)(wlds + row * 40 + c4) = p;
    }
  };

  const int KT = K >> 5;
  gload(0);
  for (int kt = 0; kt < KT; ++kt) {
    __syncthreads();   // previous step's LDS reads done
    lstore();          // commit prefetched tile (waits loadcnt here)
    __syncthreads();   // tile visible
    if (kt + 1 < KT) gload((kt + 1) << 5);  // issue next loads, no wait

    // ---- fragments + 8 WMMAs ----
    v16bf fa[4], fb[2];
#pragma unroll
    for (int mt = 0; mt < 4; ++mt) {
      const u16* ap = alds + (wm + mt * 16 + ln) * 40;  // A: m=ln, k=8h+e / 16+8h+e
      fa[mt] = ld_frag(ap, 8 * hi, 16 + 8 * hi);
    }
#pragma unroll
    for (int nt = 0; nt < 2; ++nt) {
      const u16* wp = wlds + (wn + nt * 16 + ln) * 40;  // B: n=ln, k=8h+e / 16+8h+e
      fb[nt] = ld_frag(wp, 8 * hi, 16 + 8 * hi);
    }
#pragma unroll
    for (int mt = 0; mt < 4; ++mt)
#pragma unroll
      for (int nt = 0; nt < 2; ++nt) acc[mt][nt] = wmma_bf16(fa[mt], fb[nt], acc[mt][nt]);
  }

  // ---- epilogue: bias + store (C layout: row = r + 8*hi, col = ln) ----
#pragma unroll
  for (int nt = 0; nt < 2; ++nt) {
    const int gn = n0 + wn + nt * 16 + ln;
    const float bv = bias[gn];
#pragma unroll
    for (int mt = 0; mt < 4; ++mt) {
#pragma unroll
      for (int r = 0; r < 8; ++r) {
        const int gm = m0 + wm + mt * 16 + r + 8 * hi;
        const float v = acc[mt][nt][r] + bv;
        if constexpr (OUT_BF16)
          ((u16*)Cp)[(size_t)gm * N + gn] = f2bf(v);
        else
          ((float*)Cp)[(size_t)gm * N + gn] = v;
      }
    }
  }
}

// ---------------------------------------------------------------------------
// Flash attention: scores[f,t] = k_f . q_t, causal keep t >= f, bias -1e4,
// softmax over f, out[t,c] = sum_f p[f,t] v[f,c].
// Grid: (qblock=32, h=8, b=2); block 256 threads = 8 waves.
// Wave (qg, ch): qg -> 16-query group, ch -> 128-wide half of HEAD_DIM=256.
// K/V tile for block kb+1 is prefetched into registers during compute of kb.
// ---------------------------------------------------------------------------
__global__ __launch_bounds__(256) void attn_flash_wmma(const u16* __restrict__ qkv,
                                                       u16* __restrict__ vals) {
  constexpr int S = 2048, ROW = 6144, NHD = 2048;
  __shared__ __align__(16) unsigned char smem[47616];
  u16* qlds = (u16*)smem;             // phase 0: [64][264] Q tile
  u16* klds = (u16*)smem;             // phase 1: [32][264] K tile (aliases qlds)
  u16* vtlds = (u16*)(smem + 16896);  // phase 1: [256][40] V^T tile

  const int tid = threadIdx.x;
  const int lane = tid & 31, wv = tid >> 5;
  const int ln = lane & 15, hi = lane >> 4;
  const int qg = wv >> 1, ch = wv & 1;
  const int qb = blockIdx.x, h = blockIdx.y, b = blockIdx.z;
  u16* plds = (u16*)(smem + 37376 + wv * 1280);  // per-wave [16][40] P^T tile

  const size_t base = (size_t)(b * S) * ROW + (size_t)h * 256;

  // K/V prefetch registers + helpers
  uint4 pk[4], pv4[4];
  auto kvload = [&](int f0) {
#pragma unroll
    for (int i = 0; i < 4; ++i) {
      int vec = i * 256 + tid;
      int row = vec >> 5, c8 = (vec & 31) << 3;
      pk[i] = *(const uint4*)(qkv + base + (size_t)(f0 + row) * ROW + 2048 + c8);
      pv4[i] = *(const uint4*)(qkv + base + (size_t)(f0 + row) * ROW + 4096 + c8);
    }
  };
  auto kvstore = [&]() {
#pragma unroll
    for (int i = 0; i < 4; ++i) {
      int vec = i * 256 + tid;
      int row = vec >> 5, c8 = (vec & 31) << 3;
      *(uint4*)(klds + row * 264 + c8) = pk[i];
      u32 w4[4] = {pv4[i].x, pv4[i].y, pv4[i].z, pv4[i].w};
#pragma unroll
      for (int j = 0; j < 4; ++j) {
        vtlds[(c8 + 2 * j) * 40 + row] = (u16)(w4[j] & 0xffffu);       // V^T scatter
        vtlds[(c8 + 2 * j + 1) * 40 + row] = (u16)(w4[j] >> 16);
      }
    }
  };

  kvload(0);  // overlap first K/V fetch with Q staging below

  // ---- stage Q tile [64 rows][256 c] ----
#pragma unroll
  for (int i = 0; i < 8; ++i) {
    int vec = i * 256 + tid;
    int row = vec >> 5, c8 = (vec & 31) << 3;
    uint4 a = *(const uint4*)(qkv + base + (size_t)(qb * 64 + row) * ROW + c8);
    *(uint4*)(qlds + row * 264 + c8) = a;
  }
  __syncthreads();

  const int t0 = qb * 64 + qg * 16;
  const int tg = t0 + ln;  // this lane's query index for softmax bookkeeping

  // Q fragments in registers (B operand: n = t, k = c)
  v16bf bq[8];
#pragma unroll
  for (int kc = 0; kc < 8; ++kc) {
    const u16* qp = qlds + (qg * 16 + ln) * 264 + kc * 32;
    bq[kc] = ld_frag(qp, 8 * hi, 16 + 8 * hi);
  }

  v8f oacc[8] = {};
  float m_t = -3.0e38f, l_t = 0.f;

  const int nkb = 2 * qb + 2;  // key blocks of 32 covering f <= qb*64+63
  for (int kb = 0; kb < nkb; ++kb) {
    const int f0 = kb * 32;
    __syncthreads();  // previous compute's LDS reads done (also guards Q reads)
    kvstore();        // commit prefetched K/V tile
    __syncthreads();
    if (kb + 1 < nkb) kvload(f0 + 32);  // issue next K/V loads, no wait

    // ---- scores D[f,t]: two 16x16 tiles stacked along f ----
    v8f sacc[2] = {};
#pragma unroll
    for (int kc = 0; kc < 8; ++kc) {
#pragma unroll
      for (int ft = 0; ft < 2; ++ft) {
        const u16* kp = klds + (ft * 16 + ln) * 264 + kc * 32 + 8 * hi;
        v16bf fk = ld_frag(kp, 0, 16);
        sacc[ft] = wmma_bf16(fk, bq[kc], sacc[ft]);
      }
    }

    // ---- causal bias + online softmax over f (partner lane = lane^16) ----
    float sv[2][8];
    float mloc = -3.0e38f;
#pragma unroll
    for (int ft = 0; ft < 2; ++ft)
#pragma unroll
      for (int r = 0; r < 8; ++r) {
        const int fg = f0 + ft * 16 + r + 8 * hi;
        float s = (tg < fg) ? -10000.0f : sacc[ft][r];
        sv[ft][r] = s;
        mloc = fmaxf(mloc, s);
      }
    mloc = fmaxf(mloc, __shfl_xor(mloc, 16, 32));
    const float mnew = fmaxf(m_t, mloc);
    float ps = 0.f;
#pragma unroll
    for (int ft = 0; ft < 2; ++ft)
#pragma unroll
      for (int r = 0; r < 8; ++r) {
        const float p = __expf(sv[ft][r] - mnew);
        sv[ft][r] = p;
        ps += p;
      }
    ps += __shfl_xor(ps, 16, 32);
    const float alpha = __expf(m_t - mnew);
    l_t = l_t * alpha + ps;
    m_t = mnew;

    // rescale running output (out rows use t = r + 8*hi -> fetch alpha per row)
#pragma unroll
    for (int r = 0; r < 8; ++r) {
      const float ar = __shfl(alpha, r + 8 * hi, 32);
#pragma unroll
      for (int ct = 0; ct < 8; ++ct) oacc[ct][r] *= ar;
    }

    // ---- write P^T [t][f] to per-wave LDS (pairs of consecutive f) ----
#pragma unroll
    for (int ft = 0; ft < 2; ++ft)
#pragma unroll
      for (int j = 0; j < 4; ++j) {
        const u32 pkd = pack2(f2bf(sv[ft][2 * j]), f2bf(sv[ft][2 * j + 1]));
        *(u32*)(plds + ln * 40 + ft * 16 + 8 * hi + 2 * j) = pkd;
      }

    // P as A operand (m = t, k = f)
    const u16* pr = plds + ln * 40;
    const v16bf fp = ld_frag(pr, 8 * hi, 16 + 8 * hi);

    // ---- PV: D[t,c] over this wave's 128-wide c half ----
#pragma unroll
    for (int ct = 0; ct < 8; ++ct) {
      const u16* vp = vtlds + (ch * 128 + ct * 16 + ln) * 40;
      v16bf fv = ld_frag(vp, 8 * hi, 16 + 8 * hi);
      oacc[ct] = wmma_bf16(fp, fv, oacc[ct]);
    }
  }

  // ---- normalize and store vals[b, s, h*256 + c] as bf16 ----
#pragma unroll
  for (int r = 0; r < 8; ++r) {
    const float linv = 1.0f / __shfl(l_t, r + 8 * hi, 32);
    const int sg = t0 + r + 8 * hi;
#pragma unroll
    for (int ct = 0; ct < 8; ++ct) {
      const int c = h * 256 + ch * 128 + ct * 16 + ln;
      vals[(size_t)(b * S + sg) * NHD + c] = f2bf(oacc[ct][r] * linv);
    }
  }
}

// ---------------------------------------------------------------------------
extern "C" void kernel_launch(void* const* d_in, const int* in_sizes, int n_in,
                              void* d_out, int out_size, void* d_ws, size_t ws_size,
                              hipStream_t stream) {
  (void)in_sizes; (void)n_in; (void)out_size; (void)ws_size;
  const float* x = (const float*)d_in[0];      // [2,2048,4096]
  const float* w_qkv = (const float*)d_in[1];  // [6144,4096]
  const float* b_qkv = (const float*)d_in[2];  // [6144]
  const float* w_o = (const float*)d_in[3];    // [4096,2048]
  const float* b_o = (const float*)d_in[4];    // [4096]

  u16* qkv = (u16*)d_ws;                      // [2,2048,6144] bf16 (~50 MB)
  u16* vals = qkv + (size_t)2 * 2048 * 6144;  // [2,2048,2048] bf16 (~17 MB)

  dim3 blk(256);
  // QKV projection: [4096,6144] = x[4096,4096] @ w_qkv^T + b_qkv  -> bf16 ws
  gemm_bias_wmma<false, true><<<dim3(6144 / 128, 4096 / 128), blk, 0, stream>>>(
      x, w_qkv, b_qkv, qkv, 4096, 6144, 4096);
  // Causal flash attention per (b, h, 64-query block)
  attn_flash_wmma<<<dim3(32, 8, 2), blk, 0, stream>>>(qkv, vals);
  // Output projection: [4096,4096] = vals[4096,2048] @ w_o^T + b_o -> fp32 out
  gemm_bias_wmma<true, false><<<dim3(4096 / 128, 4096 / 128), blk, 0, stream>>>(
      vals, w_o, b_o, d_out, 4096, 4096, 2048);
}
